// FeatureExtractor_65738769432679
// MI455X (gfx1250) — matrix operand order repaired and checked by reading, not compiled, not run on hardware
//
#include <hip/hip_runtime.h>
#include <math.h>

// ---------------- problem constants (from reference) ----------------
#define FRAME_LEN 512
#define HOP       256
#define NS        160000
#define BATCH     64
#define NF        257          // rfft bins
#define NFP       272          // padded to 17 * 16
#define NT        624          // (NS - FRAME_LEN)/HOP + 1 = 39 * 16 exactly
#define FT_TILES  17
#define TT_TILES  39
#define EPSF      1.1920928955078125e-7f   // float32 eps
#define TWO_PI    6.283185307179586f
#define BROW      528          // LDS B row stride in halfs (512 + 16 pad,
                               // 1056B != k*256B -> spreads LDS banks)

typedef __attribute__((ext_vector_type(16))) _Float16 v16h;
typedef __attribute__((ext_vector_type(8)))  float    v8f;

// =====================================================================
// Kernel 1: build windowed DFT basis in f16.
//   Kc[f,l] = cos(2*pi*f*l/512) * hann(l)
//   Ks[f,l] = sin(2*pi*f*l/512) * hann(l)   (i = xf . Ks; reference does
//   i = -einsum(xf, -sin*win))
// Phase reduced exactly: (f*l) mod 512 -> cosf/sinf args in [0, 2pi).
// =====================================================================
__global__ void dft_init_kernel(_Float16* __restrict__ Kc,
                                _Float16* __restrict__ Ks) {
    int idx = blockIdx.x * blockDim.x + threadIdx.x;
    if (idx >= NFP * FRAME_LEN) return;
    int f = idx >> 9;          // / 512
    int l = idx & 511;         // % 512
    float win = 0.5f * (1.0f - cosf((float)l * (TWO_PI / 512.0f)));
    int   ph  = (f * l) & 511;                 // exact phase reduction
    float ang = (float)ph * (TWO_PI / 512.0f);
    Kc[idx] = (_Float16)(cosf(ang) * win);
    Ks[idx] = (_Float16)(sinf(ang) * win);
}

// =====================================================================
// Kernel 2 (fused): WMMA STFT magnitude + per-(b,f) standardization.
//
// One 256-thread block (8 wave32) owns one (batch, 16-bin f-tile) and ALL
// 39 t-tiles: wave w handles t-tiles w, w+8, ... Each wave computes
// 16(t) x 16(f) tiles with v_wmma_f32_16x16x32_f16 (K=512, 16 steps x2).
//
// B tiles (Kc/Ks rows f0..f0+15, 16KB each) are staged into LDS ONCE per
// block with global_load_async_to_lds_b128 (ASYNCcnt), then every wave
// reads its fragments with ds loads -> 8x less VMEM for B.
//
// Per-lane partial sum/sumsq of clipped mag are tree-reduced across waves
// in LDS; feat = (clip(mag)-mean)/(std+eps) written in a second pass that
// re-reads the block's own freshly written mag lines (L2-resident).
// =====================================================================
__global__ void __launch_bounds__(256)
stft_fused_kernel(const float*    __restrict__ x,
                  const _Float16* __restrict__ Kc,
                  const _Float16* __restrict__ Ks,
                  float*          __restrict__ mag_out,
                  float*          __restrict__ feat_out) {
    __shared__ __align__(32) _Float16 sB[2 * 16 * BROW];  // [cos|sin][16 f][BROW]
    __shared__ float psum[8][32];
    __shared__ float psq [8][32];
    __shared__ float sstat[2][16];                        // mean, 1/(std+eps)

    const int tid  = threadIdx.x;
    const int lane = tid & 31;
    const int wave = tid >> 5;
    const int f0   = blockIdx.x * 16;
    const int b    = blockIdx.y;

    // ---------- async-stage Kc/Ks tiles into LDS (16 rows x 1KB each) ----------
    {
        const unsigned lds0 = (unsigned)(size_t)(&sB[0]);  // flat LDS addr: low 32b = LDS offset
#pragma unroll
        for (int mat = 0; mat < 2; ++mat) {
            const _Float16* gsrc = (mat == 0 ? Kc : Ks) + (size_t)f0 * FRAME_LEN;
#pragma unroll
            for (int it = 0; it < 4; ++it) {
                int c   = tid + it * 256;          // 1024 x 16B chunks per matrix
                int row = c >> 6;                  // 64 chunks per f-row
                int col = (c & 63) * 8;            // halfs
                unsigned lds = lds0 + (unsigned)(((mat * 16 + row) * BROW + col) * 2);
                unsigned long long g =
                    (unsigned long long)(size_t)(gsrc + (size_t)row * FRAME_LEN + col);
                asm volatile("global_load_async_to_lds_b128 %0, %1, off"
                             :: "v"(lds), "v"(g) : "memory");
            }
        }
        asm volatile("s_wait_asynccnt 0" ::: "memory");
    }
    __syncthreads();

    // ---------- per-lane fragment coordinates (m/n per ISA; K contiguous per lane half) ----------
    const int m     = lane & 15;       // A row (t in tile) == B/D col (f in tile)
    const int hi    = lane >> 4;       // lane half
    const int khalf = hi * 16;         // A: K group
    const int kbase = hi * 16;         // B: K group
    const int f     = f0 + m;

    const _Float16* sBc = &sB[(size_t)m * BROW];
    const _Float16* sBs = &sB[(size_t)(16 + m) * BROW];

    float lsum = 0.0f, lsq = 0.0f;
    const size_t obase = (size_t)b * NF * NT + (size_t)f * NT;

    for (int tt = wave; tt < TT_TILES; tt += 8) {          // wave-uniform loop
        const float* xrow = x + (size_t)b * NS + (size_t)(tt * 16 + m) * HOP;
        v8f accr = {};
        v8f acci = {};
        for (int kb = 0; kb < FRAME_LEN; kb += 32) {
            // A fragment: lanes 0-15 K=kb+{0..15}, lanes 16-31 +16
            const float4* pa0 = (const float4*)(xrow + kb + khalf);
            const float4* pa1 = (const float4*)(xrow + kb + khalf + 8);
            float4 a0 = pa0[0], a1 = pa0[1];
            float4 a2 = pa1[0], a3 = pa1[1];
            if (kb + 32 < FRAME_LEN)
                __builtin_prefetch(xrow + kb + 32, 0, 3);  // global_prefetch_b8

            v16h A;
            A[0]  = (_Float16)a0.x; A[1]  = (_Float16)a0.y;
            A[2]  = (_Float16)a0.z; A[3]  = (_Float16)a0.w;
            A[4]  = (_Float16)a1.x; A[5]  = (_Float16)a1.y;
            A[6]  = (_Float16)a1.z; A[7]  = (_Float16)a1.w;
            A[8]  = (_Float16)a2.x; A[9]  = (_Float16)a2.y;
            A[10] = (_Float16)a2.z; A[11] = (_Float16)a2.w;
            A[12] = (_Float16)a3.x; A[13] = (_Float16)a3.y;
            A[14] = (_Float16)a3.z; A[15] = (_Float16)a3.w;

            // B fragments from LDS: lane = col n, 16 sequential K (2x ds b128)
            v16h Bc = *(const v16h*)(sBc + kb + kbase);
            v16h Bs = *(const v16h*)(sBs + kb + kbase);

            accr = __builtin_amdgcn_wmma_f32_16x16x32_f16(
                       false, A, false, Bc, (short)0, accr, false, false);
            acci = __builtin_amdgcn_wmma_f32_16x16x32_f16(
                       false, A, false, Bs, (short)0, acci, false, false);
        }

        // D element v -> t = tt*16 + v + 8*hi, f = f0 + (lane&15)
        const int t0 = tt * 16 + hi * 8;
        float mg[8];
#pragma unroll
        for (int v = 0; v < 8; ++v) {
            float r = accr[v], ii = acci[v];
            mg[v] = sqrtf(r * r + ii * ii);            // raw mag
            float c = fmaxf(mg[v], EPSF);              // clipped for stats
            lsum += c; lsq += c * c;
        }
        if (f < NF) {
#pragma unroll
            for (int v = 0; v < 8; ++v) mag_out[obase + t0 + v] = mg[v];
        }
    }

    // ---------- cross-wave mean/std (ddof=1) per f row ----------
    psum[wave][lane] = lsum;
    psq [wave][lane] = lsq;
    __syncthreads();

    if (tid < 16) {
        float s = 0.0f, q = 0.0f;
#pragma unroll
        for (int w = 0; w < 8; ++w) {
            s += psum[w][tid] + psum[w][tid + 16];   // both lane-halves = same f
            q += psq [w][tid] + psq [w][tid + 16];
        }
        float mean = s * (1.0f / (float)NT);
        float var  = (q - (float)NT * mean * mean) * (1.0f / (float)(NT - 1));
        sstat[0][tid] = mean;
        sstat[1][tid] = 1.0f / (sqrtf(fmaxf(var, 0.0f)) + EPSF);
    }
    __syncthreads();

    const float mean = sstat[0][m];
    const float inv  = sstat[1][m];

    // ---------- feat pass: re-read own mag lines (same-wave order, L2 hot) ----------
    if (f < NF) {
        for (int tt = wave; tt < TT_TILES; tt += 8) {
            const int t0 = tt * 16 + hi * 8;
#pragma unroll
            for (int v = 0; v < 8; ++v) {
                float c = fmaxf(mag_out[obase + t0 + v], EPSF);
                feat_out[obase + t0 + v] = (c - mean) * inv;
            }
        }
    }
}

// =====================================================================
// launcher
// =====================================================================
extern "C" void kernel_launch(void* const* d_in, const int* in_sizes, int n_in,
                              void* d_out, int out_size, void* d_ws, size_t ws_size,
                              hipStream_t stream) {
    (void)in_sizes; (void)n_in; (void)out_size; (void)ws_size;

    const float* x   = (const float*)d_in[0];
    float*       out = (float*)d_out;                  // [mag | feat], each B*F*T
    _Float16*    Kc  = (_Float16*)d_ws;                // NFP*FRAME_LEN f16
    _Float16*    Ks  = Kc + (size_t)NFP * FRAME_LEN;   // NFP*FRAME_LEN f16

    // 1) basis init (tiny: 272*512 elements)
    {
        int n = NFP * FRAME_LEN;
        dft_init_kernel<<<(n + 255) / 256, 256, 0, stream>>>(Kc, Ks);
    }
    // 2) fused WMMA STFT magnitude + standardization
    {
        float* feat = out + (size_t)BATCH * NF * NT;
        dim3 block(256);
        dim3 grid(FT_TILES, BATCH);                    // (17, 64) blocks
        stft_fused_kernel<<<grid, block, 0, stream>>>(x, Kc, Ks, out, feat);
    }
}
